// CausalMultiHeadSelfAttention_1142461301094
// MI455X (gfx1250) — matrix operand order repaired and checked by reading, not compiled, not run on hardware
//
#include <hip/hip_runtime.h>
#include <hip/hip_bf16.h>

// ---------------------------------------------------------------------------
// Causal multi-head self-attention for MI455X (gfx1250, wave32, WMMA).
// All GEMMs use v_wmma_f32_16x16x32_bf16 (bf16 in, fp32 accumulate).
// Softmax / RoPE in fp32 VALU (v_exp_f32 co-executes with XDL WMMA).
// GEMM kernels use 32x64 register tiles per wave: 8 WMMAs per 12 b128 loads.
// ---------------------------------------------------------------------------

typedef __attribute__((ext_vector_type(16))) __bf16 v16bf;
typedef __attribute__((ext_vector_type(8)))  float  v8f;

#define BATCH   2
#define S_LEN   2048
#define DMODEL  1024
#define NHEAD   16
#define DK      64

#define WMMA_BF16(A, B, C) \
    __builtin_amdgcn_wmma_f32_16x16x32_bf16(false, (A), false, (B), (short)0, (C), false, false)

// ---- helpers --------------------------------------------------------------

__device__ __forceinline__ unsigned short f2bf(float f) {
    unsigned int u = __float_as_uint(f);
    u += 0x7FFFu + ((u >> 16) & 1u);               // round-to-nearest-even
    return (unsigned short)(u >> 16);
}

__device__ __forceinline__ unsigned int ld32(const unsigned short* p) {
    return *reinterpret_cast<const unsigned int*>(p);
}

// A-matrix 16x32 bf16 fragment (ISA 7.12.2): lane L -> row = L%16,
// VGPR v in 0..3: K = 8*(L/16) + 2v {+0,+1}; v in 4..7: K = 16 + 8*(L/16) + 2(v-4).
// `row` must already point at this lane's row (K-contiguous memory).
__device__ __forceinline__ v16bf load_a_frag(const unsigned short* row, int k0, int half) {
    union { unsigned int u[8]; v16bf v; } f;
#pragma unroll
    for (int i = 0; i < 4; ++i) f.u[i]     = ld32(row + k0 +      half * 8 + 2 * i);
#pragma unroll
    for (int i = 0; i < 4; ++i) f.u[4 + i] = ld32(row + k0 + 16 + half * 8 + 2 * i);
    return f.v;
}

// B-matrix 32x16 bf16 fragment: lane L -> col n = L%16,
// VGPR v in 0..3: K = 8*(L/16) + 2v {+0,+1}; v in 4..7: K = 16 + 8*(L/16) + 2(v-4).
// `row` must point at this lane's n-column data laid out K-contiguously.
__device__ __forceinline__ v16bf load_b_frag(const unsigned short* row, int k0, int half) {
    union { unsigned int u[8]; v16bf v; } f;
#pragma unroll
    for (int i = 0; i < 4; ++i) f.u[i]     = ld32(row + k0 +      half * 8 + 2 * i);
#pragma unroll
    for (int i = 0; i < 4; ++i) f.u[4 + i] = ld32(row + k0 + 16 + half * 8 + 2 * i);
    return f.v;
}

// ---- kernel 0: fp32 -> bf16 convert ---------------------------------------

__global__ void cvt_bf16_kernel(const float* __restrict__ in,
                                unsigned short* __restrict__ out, int n) {
    int i = blockIdx.x * blockDim.x + threadIdx.x;
    if (i < n) out[i] = f2bf(in[i]);
}

// ---- kernel 1: QKV projection + fused RoPE --------------------------------
// C[m,e] = sum_d x[m,d] * qkv_w[e,d].  One wave = 32 rows x 64 cols,
// and a 64-col strip == one (matrix, head): strip/16 in {q,k,v}, strip%16 = head.

__global__ __launch_bounds__(256) void qkv_rope_kernel(
    const unsigned short* __restrict__ xb,     // [B*S, D] bf16
    const unsigned short* __restrict__ wqkv,   // [3D, D] bf16
    const int* __restrict__ pos,               // [S]
    unsigned short* __restrict__ qbuf,         // [B*H, S, DK] bf16 (RoPE'd)
    unsigned short* __restrict__ kbuf,         // [B*H, S, DK] bf16 (RoPE'd)
    unsigned short* __restrict__ vT) {         // [B*H, DK, S] bf16 (transposed)
    const int lane = threadIdx.x & 31;
    const int wave = threadIdx.x >> 5;
    const int task = blockIdx.x * 8 + wave;    // 128 row tiles * 48 strips
    const int mtile = task / 48;               // 32-row tiles
    const int strip = task % 48;
    const int half = lane >> 4;
    const int ln   = lane & 15;

    v8f acc[2][4] = {};
    const unsigned short* arow0 = xb + (size_t)(mtile * 32 + ln) * DMODEL;
    const unsigned short* arow1 = arow0 + (size_t)16 * DMODEL;
    for (int k0 = 0; k0 < DMODEL; k0 += 32) {
        v16bf a0 = load_a_frag(arow0, k0, half);
        v16bf a1 = load_a_frag(arow1, k0, half);
#pragma unroll
        for (int nf = 0; nf < 4; ++nf) {
            const unsigned short* brow =
                wqkv + (size_t)(strip * 64 + nf * 16 + ln) * DMODEL;
            v16bf b = load_b_frag(brow, k0, half);
            acc[0][nf] = WMMA_BF16(a0, b, acc[0][nf]);
            acc[1][nf] = WMMA_BF16(a1, b, acc[1][nf]);
        }
    }

    const int mat = strip >> 4;   // 0=q, 1=k, 2=v
    const int h   = strip & 15;

    if (mat < 2) {
        unsigned short* dst = (mat == 0) ? qbuf : kbuf;
#pragma unroll
        for (int mt = 0; mt < 2; ++mt) {
#pragma unroll
            for (int nf = 0; nf < 4; ++nf) {
                const int d = nf * 16 + ln;                  // head dim 0..63
                const int i = d >> 1;
                // inv_freq = 10000^(-i/32) = exp(-i * ln(10000)/32)
                const float invf =
                    __expf(-(float)i * (9.210340371976184f / 32.0f));
#pragma unroll
                for (int vv = 0; vv < 8; ++vv) {
                    const int row = mtile * 32 + mt * 16 + vv + 8 * half;
                    const int s = row & (S_LEN - 1);
                    const int b = row >> 11;
                    const float ang = (float)pos[s] * invf;
                    float c, sn;
                    __sincosf(ang, &c, &sn);
                    const float x  = acc[mt][nf][vv];
                    const float xp = __shfl_xor(x, 1, 32);    // even<->odd dim
                    const float r = (d & 1) ? (xp * sn + x * c)   // r_odd
                                            : (x * c - xp * sn);  // r_even
                    dst[((size_t)((b * NHEAD + h) * S_LEN + s)) * DK + d] =
                        f2bf(r);
                }
            }
        }
    } else {
#pragma unroll
        for (int mt = 0; mt < 2; ++mt) {
#pragma unroll
            for (int nf = 0; nf < 4; ++nf) {
                const int d = nf * 16 + ln;
#pragma unroll
                for (int vv = 0; vv < 8; ++vv) {
                    const int row = mtile * 32 + mt * 16 + vv + 8 * half;
                    const int s = row & (S_LEN - 1);
                    const int b = row >> 11;
                    vT[((size_t)((b * NHEAD + h) * DK + d)) * S_LEN + s] =
                        f2bf(acc[mt][nf][vv]);
                }
            }
        }
    }
}

// ---- kernel 2: flash attention (causal, online softmax) -------------------
// One wave per (b, h, 16-query tile). Keys processed 32 at a time:
// 4 score WMMAs + 4 P*V WMMAs per iteration.

__global__ __launch_bounds__(128) void attn_kernel(
    const unsigned short* __restrict__ qbuf,   // [B*H, S, DK]
    const unsigned short* __restrict__ kbuf,   // [B*H, S, DK]
    const unsigned short* __restrict__ vT,     // [B*H, DK, S]
    unsigned short* __restrict__ obuf) {       // [B, S, D] bf16
    __shared__ unsigned short plds[4][16][32]; // per-wave P staging tile

    const int lane = threadIdx.x & 31;
    const int wave = threadIdx.x >> 5;
    const int task = blockIdx.x * 4 + wave;    // 32 bh * 128 qtiles
    const int bh    = task >> 7;
    const int qtile = task & 127;
    const int q0   = qtile * 16;
    const int half = lane >> 4;
    const int ln   = lane & 15;

    // Q fragments for the whole d_k = 64 (two 16x32 A-frags), loaded once.
    const unsigned short* qrow = qbuf + ((size_t)bh * S_LEN + q0 + ln) * DK;
    const v16bf aq0 = load_a_frag(qrow, 0, half);
    const v16bf aq1 = load_a_frag(qrow, 32, half);

    v8f o[4] = {};
    float mrow[8], lrow[8];
#pragma unroll
    for (int i = 0; i < 8; ++i) { mrow[i] = -3.0e38f; lrow[i] = 0.0f; }

    for (int j0 = 0; j0 < q0 + 16; j0 += 32) {
        // ---- scores: S = Q @ K^T (two 16-key sub-tiles) ----
        v8f sa[2];
#pragma unroll
        for (int nf = 0; nf < 2; ++nf) {
            const unsigned short* krow =
                kbuf + ((size_t)bh * S_LEN + j0 + nf * 16 + ln) * DK;
            v8f s = {};
            s = WMMA_BF16(aq0, load_b_frag(krow, 0, half), s);
            s = WMMA_BF16(aq1, load_b_frag(krow, 32, half), s);
            sa[nf] = s;
        }

        // ---- causal mask + online softmax (rows live in 16-lane halves) ----
        float p0[8], p1[8];
#pragma unroll
        for (int vv = 0; vv < 8; ++vv) {
            const int qrow_g = q0 + vv + 8 * half;
            float s0 = sa[0][vv] * 0.125f;           // 1/sqrt(64)
            float s1 = sa[1][vv] * 0.125f;
            if (j0 + ln > qrow_g)      s0 = -3.0e38f;
            if (j0 + 16 + ln > qrow_g) s1 = -3.0e38f;

            float t = fmaxf(s0, s1);
            t = fmaxf(t, __shfl_xor(t, 1, 32));
            t = fmaxf(t, __shfl_xor(t, 2, 32));
            t = fmaxf(t, __shfl_xor(t, 4, 32));
            t = fmaxf(t, __shfl_xor(t, 8, 32));
            const float mnew  = fmaxf(mrow[vv], t);
            const float scale = __expf(mrow[vv] - mnew);
            mrow[vv] = mnew;

            const float e0 = __expf(s0 - mnew);
            const float e1 = __expf(s1 - mnew);
            float rs = e0 + e1;
            rs += __shfl_xor(rs, 1, 32);
            rs += __shfl_xor(rs, 2, 32);
            rs += __shfl_xor(rs, 4, 32);
            rs += __shfl_xor(rs, 8, 32);
            lrow[vv] = lrow[vv] * scale + rs;
#pragma unroll
            for (int c = 0; c < 4; ++c) o[c][vv] *= scale;
            p0[vv] = e0;
            p1[vv] = e1;
        }

        // ---- P: C-layout -> LDS -> A-layout (same wave, DS in-order) ----
#pragma unroll
        for (int vv = 0; vv < 8; ++vv) {
            const int m = vv + 8 * half;
            plds[wave][m][ln]      = f2bf(p0[vv]);
            plds[wave][m][16 + ln] = f2bf(p1[vv]);
        }
        asm volatile("s_wait_dscnt 0" ::: "memory");

        union { unsigned int u[8]; v16bf v; } pf;
        const unsigned int* prow =
            reinterpret_cast<const unsigned int*>(&plds[wave][ln][0]);
#pragma unroll
        for (int i = 0; i < 4; ++i) pf.u[i]     = prow[half * 4 + i];
#pragma unroll
        for (int i = 0; i < 4; ++i) pf.u[4 + i] = prow[8 + half * 4 + i];

        // ---- O += P @ V  (V transposed so pairs are contiguous) ----
#pragma unroll
        for (int c = 0; c < 4; ++c) {
            const unsigned short* vrow =
                vT + ((size_t)bh * DK + c * 16 + ln) * S_LEN + j0;
            o[c] = WMMA_BF16(pf.v, load_b_frag(vrow, 0, half), o[c]);
        }
    }

    // ---- normalize and write out [B,S,D] ----
    const int b = bh >> 4, h = bh & 15;
#pragma unroll
    for (int vv = 0; vv < 8; ++vv) {
        const float inv = 1.0f / lrow[vv];
        const int row = q0 + vv + 8 * half;
#pragma unroll
        for (int c = 0; c < 4; ++c) {
            const int col = h * 64 + c * 16 + ln;
            obuf[((size_t)(b * S_LEN + row)) * DMODEL + col] =
                f2bf(o[c][vv] * inv);
        }
    }
}

// ---- kernel 3: output projection, fp32 result -----------------------------

__global__ __launch_bounds__(256) void out_proj_kernel(
    const unsigned short* __restrict__ ob,     // [B*S, D] bf16
    const unsigned short* __restrict__ wout,   // [D, D] bf16
    float* __restrict__ out) {                 // [B*S, D] fp32
    const int lane = threadIdx.x & 31;
    const int wave = threadIdx.x >> 5;
    const int task = blockIdx.x * 8 + wave;    // 128 row tiles * 16 strips
    const int mtile = task >> 4;               // 32-row tiles
    const int strip = task & 15;
    const int half = lane >> 4;
    const int ln   = lane & 15;

    v8f acc[2][4] = {};
    const unsigned short* arow0 = ob + (size_t)(mtile * 32 + ln) * DMODEL;
    const unsigned short* arow1 = arow0 + (size_t)16 * DMODEL;
    for (int k0 = 0; k0 < DMODEL; k0 += 32) {
        v16bf a0 = load_a_frag(arow0, k0, half);
        v16bf a1 = load_a_frag(arow1, k0, half);
#pragma unroll
        for (int nf = 0; nf < 4; ++nf) {
            const unsigned short* brow =
                wout + (size_t)(strip * 64 + nf * 16 + ln) * DMODEL;
            v16bf b = load_b_frag(brow, k0, half);
            acc[0][nf] = WMMA_BF16(a0, b, acc[0][nf]);
            acc[1][nf] = WMMA_BF16(a1, b, acc[1][nf]);
        }
    }
#pragma unroll
    for (int mt = 0; mt < 2; ++mt)
#pragma unroll
        for (int nf = 0; nf < 4; ++nf)
#pragma unroll
            for (int vv = 0; vv < 8; ++vv)
                out[(size_t)(mtile * 32 + mt * 16 + vv + 8 * half) * DMODEL +
                    strip * 64 + nf * 16 + ln] = acc[mt][nf][vv];
}

// ---- host launch ----------------------------------------------------------

extern "C" void kernel_launch(void* const* d_in, const int* in_sizes, int n_in,
                              void* d_out, int out_size, void* d_ws, size_t ws_size,
                              hipStream_t stream) {
    const float* x     = (const float*)d_in[0];   // [B,S,D]
    const float* qkv_w = (const float*)d_in[1];   // [3D,D]
    const float* out_w = (const float*)d_in[2];   // [D,D]
    const int*   pos   = (const int*)d_in[3];     // [S]

    unsigned short* ws = (unsigned short*)d_ws;
    size_t off = 0;
    unsigned short* xb   = ws + off; off += (size_t)BATCH * S_LEN * DMODEL;
    unsigned short* wqkv = ws + off; off += (size_t)3 * DMODEL * DMODEL;
    unsigned short* wout = ws + off; off += (size_t)DMODEL * DMODEL;
    unsigned short* qb   = ws + off; off += (size_t)BATCH * NHEAD * S_LEN * DK;
    unsigned short* kb   = ws + off; off += (size_t)BATCH * NHEAD * S_LEN * DK;
    unsigned short* vT   = ws + off; off += (size_t)BATCH * NHEAD * DK * S_LEN;
    unsigned short* obuf = ws + off; off += (size_t)BATCH * S_LEN * DMODEL;

    const int n1 = BATCH * S_LEN * DMODEL;
    const int n2 = 3 * DMODEL * DMODEL;
    const int n3 = DMODEL * DMODEL;
    cvt_bf16_kernel<<<(n1 + 255) / 256, 256, 0, stream>>>(x, xb, n1);
    cvt_bf16_kernel<<<(n2 + 255) / 256, 256, 0, stream>>>(qkv_w, wqkv, n2);
    cvt_bf16_kernel<<<(n3 + 255) / 256, 256, 0, stream>>>(out_w, wout, n3);

    // 128 row-tiles * 48 strips = 6144 wave-tasks, 8 waves/block
    qkv_rope_kernel<<<768, 256, 0, stream>>>(xb, wqkv, pos, qb, kb, vT);
    // 32 (b,h) * 128 q-tiles = 4096 wave-tasks, 4 waves/block
    attn_kernel<<<1024, 128, 0, stream>>>(qb, kb, vT, obuf);
    // 128 row-tiles * 16 strips = 2048 wave-tasks, 8 waves/block
    out_proj_kernel<<<256, 256, 0, stream>>>(obuf, wout, (float*)d_out);
}
